// CausalSelfAttention_55207509623141
// MI455X (gfx1250) — matrix operand order repaired and checked by reading, not compiled, not run on hardware
//
#include <hip/hip_runtime.h>
#include <hip/hip_bf16.h>
#include <math.h>

// CDNA5 / gfx1250 causal self-attention.
// Split-precision f16 WMMA (V_WMMA_F32_16X16X32_F16, Ootomo-style hi/lo):
//   D += Ahi*Bhi + Ahi*Blo + Alo*Bhi   (fp32 accumulate, ~fp32 fidelity)
// Phase 0: split/transpose prep (x, W_attn^T, W_proj^T -> f16 hi/lo)
// Phase 1: QKV gemm -> head-major Q,K (hi/lo) and transposed V (hi/lo)
// Phase 2: flash attention, 32-key tiles, online softmax
// Phase 3: projection gemm -> fp32 output

#define D_MODEL 768
#define N_HEAD  12
#define SEQ_LEN 2048
#define BATCH   4
#define HD      64
#define N_QKV   (3 * D_MODEL)

typedef _Float16 half_t;
typedef half_t v8h  __attribute__((ext_vector_type(8)));
typedef half_t v16h __attribute__((ext_vector_type(16)));
typedef float  v8f  __attribute__((ext_vector_type(8)));

__device__ __forceinline__ v8f wmma16(v16h a, v16h b, v8f c) {
  // 8 args: (neg_a, A, neg_b, B, c_mod, C, reuse_a, reuse_b)
  return __builtin_amdgcn_wmma_f32_16x16x32_f16(false, a, false, b, (short)0,
                                                c, false, false);
}

// Build a v16h fragment from two contiguous 16-byte chunks.
__device__ __forceinline__ v16h ldfrag(const half_t* p0, const half_t* p1) {
  v8h a = *(const v8h*)p0;
  v8h b = *(const v8h*)p1;
  v16h r;
#pragma unroll
  for (int i = 0; i < 8; ++i) {
    r[i] = a[i];
    r[i + 8] = b[i];
  }
  return r;
}

__device__ __forceinline__ void split32(float x, half_t* h, half_t* l) {
  half_t hi = (half_t)x;
  *h = hi;
  *l = (half_t)(x - (float)hi);
}

// ---------------------------------------------------------------------------
// Prep: elementwise split of x (row-major kept).
// ---------------------------------------------------------------------------
__global__ void split_kernel(const float* __restrict__ src,
                             half_t* __restrict__ oh, half_t* __restrict__ ol,
                             long n) {
  long stride = (long)gridDim.x * blockDim.x;
  for (long i = (long)blockIdx.x * blockDim.x + threadIdx.x; i < n;
       i += stride) {
    split32(src[i], &oh[i], &ol[i]);
  }
}

// Prep: transpose + split W[K][N] -> out[N][K] (hi/lo).
__global__ void transpose_split_kernel(const float* __restrict__ W,
                                       half_t* __restrict__ oh,
                                       half_t* __restrict__ ol, int K, int N) {
  long total = (long)K * N;
  long stride = (long)gridDim.x * blockDim.x;
  for (long i = (long)blockIdx.x * blockDim.x + threadIdx.x; i < total;
       i += stride) {
    int n = (int)(i / K);
    int k = (int)(i % K);
    split32(W[(size_t)k * N + n], &oh[i], &ol[i]);
  }
}

// ---------------------------------------------------------------------------
// Phase 1: qkv = x @ W_attn + b_attn.
// A = x hi/lo [8192 x 768], B = W_attn^T hi/lo [2304 x 768] (contiguous K).
// Wave: 64 rows x 16 cols strip.  Q,K stored [B,H,T,hd]; V stored [B,H,hd,T].
// ---------------------------------------------------------------------------
__global__ __launch_bounds__(128) void qkv_kernel(
    const half_t* __restrict__ xh, const half_t* __restrict__ xl,
    const half_t* __restrict__ wth, const half_t* __restrict__ wtl,
    const float* __restrict__ bias, half_t* __restrict__ qh,
    half_t* __restrict__ ql, half_t* __restrict__ kh,
    half_t* __restrict__ kl, half_t* __restrict__ vth,
    half_t* __restrict__ vtl) {
  const int lane = threadIdx.x & 31;
  const int half = lane >> 4;
  const int lq = lane & 15;
  const int wave = blockIdx.x * (blockDim.x >> 5) + (threadIdx.x >> 5);
  const int nTilesN = N_QKV / 16;  // 144
  const int nt = wave % nTilesN;
  const int ms = wave / nTilesN;
  const int m0 = ms * 64;
  const int n0 = nt * 16;

  const int n = n0 + lq;
  const int which = n / D_MODEL;
  const int c = n % D_MODEL;
  const int h = c >> 6;
  const int d = c & 63;
  const float bn = bias[n];
  half_t* __restrict__ dsth = (which == 0) ? qh : ((which == 1) ? kh : vth);
  half_t* __restrict__ dstl = (which == 0) ? ql : ((which == 1) ? kl : vtl);

  v8f acc[4] = {};
  for (int k0 = 0; k0 < D_MODEL; k0 += 32) {
    const half_t* bp_h = wth + (size_t)n * D_MODEL + k0 + 16 * half;
    const half_t* bp_l = wtl + (size_t)n * D_MODEL + k0 + 16 * half;
    v16h bh = ldfrag(bp_h, bp_h + 8);
    v16h bl = ldfrag(bp_l, bp_l + 8);
#pragma unroll
    for (int s = 0; s < 4; ++s) {
      const half_t* ap_h =
          xh + (size_t)(m0 + 16 * s + lq) * D_MODEL + k0 + 16 * half;
      const half_t* ap_l =
          xl + (size_t)(m0 + 16 * s + lq) * D_MODEL + k0 + 16 * half;
      v16h ah = ldfrag(ap_h, ap_h + 8);
      v16h al = ldfrag(ap_l, ap_l + 8);
      acc[s] = wmma16(ah, bh, acc[s]);
      acc[s] = wmma16(ah, bl, acc[s]);
      acc[s] = wmma16(al, bh, acc[s]);
    }
  }

#pragma unroll
  for (int s = 0; s < 4; ++s) {
#pragma unroll
    for (int r = 0; r < 8; ++r) {
      const int m = m0 + 16 * s + r + 8 * half;  // global row b*T + t
      const int bb = m >> 11;
      const int t = m & (SEQ_LEN - 1);
      const size_t head_off = (size_t)((bb * N_HEAD + h) * SEQ_LEN) * HD;
      const size_t addr = (which == 2)
                              ? head_off + (size_t)d * SEQ_LEN + t  // V^T
                              : head_off + (size_t)t * HD + d;      // Q, K
      split32(acc[s][r] + bn, &dsth[addr], &dstl[addr]);
    }
  }
}

// ---------------------------------------------------------------------------
// Phase 2: flash attention.  One wave per (b, h, 16-query tile).
// 32-key tiles: S = 2 x (16x16) WMMA tiles, P staged hi/lo through LDS.
// ---------------------------------------------------------------------------
#define FA_WAVES 4
__global__ __launch_bounds__(32 * FA_WAVES) void attn_kernel(
    const half_t* __restrict__ qh, const half_t* __restrict__ ql,
    const half_t* __restrict__ kh, const half_t* __restrict__ kl,
    const half_t* __restrict__ vth, const half_t* __restrict__ vtl,
    half_t* __restrict__ yh, half_t* __restrict__ yl) {
  __shared__ __align__(16) half_t ldsP[FA_WAVES][2][16 * 32];  // hi/lo

  const int lane = threadIdx.x & 31;
  const int half = lane >> 4;
  const int lq = lane & 15;
  const int w = threadIdx.x >> 5;
  const int wave = blockIdx.x * FA_WAVES + w;

  const int nQT = SEQ_LEN / 16;  // 128
  const int qt = wave & (nQT - 1);
  const int hh = (wave >> 7) % N_HEAD;
  const int bb = wave / (nQT * N_HEAD);
  const int q0 = qt * 16;

  const size_t head_off = (size_t)((bb * N_HEAD + hh) * SEQ_LEN) * HD;
  const half_t* __restrict__ Kh = kh + head_off;
  const half_t* __restrict__ Kl = kl + head_off;
  const half_t* __restrict__ Vh = vth + head_off;  // [hd][T]
  const half_t* __restrict__ Vl = vtl + head_off;

  half_t* p_h = ldsP[w][0];
  half_t* p_l = ldsP[w][1];

  // Preload Q fragments for this 16-query tile (row = q0+lq, K chunks 0 and 32)
  v16h qfh[2], qfl[2];
  {
    const half_t* qr_h = qh + head_off + (size_t)(q0 + lq) * HD;
    const half_t* qr_l = ql + head_off + (size_t)(q0 + lq) * HD;
#pragma unroll
    for (int cch = 0; cch < 2; ++cch) {
      const half_t* ph = qr_h + 32 * cch + 16 * half;
      const half_t* pl = qr_l + 32 * cch + 16 * half;
      qfh[cch] = ldfrag(ph, ph + 8);
      qfl[cch] = ldfrag(pl, pl + 8);
    }
  }

  v8f o[4] = {};
  float mrow[8], lrow[8];
#pragma unroll
  for (int r = 0; r < 8; ++r) {
    mrow[r] = -__builtin_inff();
    lrow[r] = 0.0f;
  }

  const int kt_max = (q0 + 15) >> 5;
  for (int kt = 0; kt <= kt_max; ++kt) {
    const int kbase = kt * 32;

    if (kt < kt_max) {  // stream-prefetch next key/value tiles
      __builtin_prefetch(Kh + (size_t)(kbase + 32 + lq) * HD, 0, 1);
      __builtin_prefetch(Vh + (size_t)lq * SEQ_LEN + kbase + 32, 0, 1);
    }

    // S = Q @ K^T : two 16x16 C tiles (keys kbase..+15, kbase+16..+31)
    v8f sacc[2] = {};
#pragma unroll
    for (int tkt = 0; tkt < 2; ++tkt) {
      const int key = kbase + 16 * tkt + lq;
#pragma unroll
      for (int cch = 0; cch < 2; ++cch) {
        const half_t* kp_h = Kh + (size_t)key * HD + 32 * cch + 16 * half;
        const half_t* kp_l = Kl + (size_t)key * HD + 32 * cch + 16 * half;
        v16h bh = ldfrag(kp_h, kp_h + 8);
        v16h bl = ldfrag(kp_l, kp_l + 8);
        sacc[tkt] = wmma16(qfh[cch], bh, sacc[tkt]);
        sacc[tkt] = wmma16(qfh[cch], bl, sacc[tkt]);
        sacc[tkt] = wmma16(qfl[cch], bh, sacc[tkt]);
      }
    }

    // scale + causal mask + online softmax across 32 keys
#pragma unroll
    for (int r = 0; r < 8; ++r) {
      const int qi = q0 + r + 8 * half;
      float sv0 = sacc[0][r] * 0.125f;  // 1/sqrt(64)
      float sv1 = sacc[1][r] * 0.125f;
      if (kbase + lq > qi) sv0 = -__builtin_inff();
      if (kbase + 16 + lq > qi) sv1 = -__builtin_inff();

      float rm = fmaxf(sv0, sv1);
      rm = fmaxf(rm, __shfl_xor(rm, 1, 32));
      rm = fmaxf(rm, __shfl_xor(rm, 2, 32));
      rm = fmaxf(rm, __shfl_xor(rm, 4, 32));
      rm = fmaxf(rm, __shfl_xor(rm, 8, 32));

      const float mnew = fmaxf(mrow[r], rm);
      const float alpha = __expf(mrow[r] - mnew);
      const float p0 = __expf(sv0 - mnew);
      const float p1 = __expf(sv1 - mnew);

      float rs = p0 + p1;
      rs += __shfl_xor(rs, 1, 32);
      rs += __shfl_xor(rs, 2, 32);
      rs += __shfl_xor(rs, 4, 32);
      rs += __shfl_xor(rs, 8, 32);

      lrow[r] = lrow[r] * alpha + rs;
      mrow[r] = mnew;
      o[0][r] *= alpha;
      o[1][r] *= alpha;
      o[2][r] *= alpha;
      o[3][r] *= alpha;

      const int prow = (r + 8 * half) * 32;
      split32(p0, &p_h[prow + lq], &p_l[prow + lq]);
      split32(p1, &p_h[prow + 16 + lq], &p_l[prow + 16 + lq]);
    }
    asm volatile("s_wait_dscnt 0x0" ::: "memory");

    // A-fragments of P (16x32, single K=32 chunk)
    const half_t* pr_h = p_h + lq * 32 + 16 * half;
    const half_t* pr_l = p_l + lq * 32 + 16 * half;
    v16h afh = ldfrag(pr_h, pr_h + 8);
    v16h afl = ldfrag(pr_l, pr_l + 8);

    // O += P @ V : V^T rows are contiguous along keys
#pragma unroll
    for (int s2 = 0; s2 < 4; ++s2) {
      const half_t* vp_h =
          Vh + (size_t)(16 * s2 + lq) * SEQ_LEN + kbase + 16 * half;
      const half_t* vp_l =
          Vl + (size_t)(16 * s2 + lq) * SEQ_LEN + kbase + 16 * half;
      v16h bh = ldfrag(vp_h, vp_h + 8);
      v16h bl = ldfrag(vp_l, vp_l + 8);
      o[s2] = wmma16(afh, bh, o[s2]);
      o[s2] = wmma16(afh, bl, o[s2]);
      o[s2] = wmma16(afl, bh, o[s2]);
    }
    asm volatile("s_wait_dscnt 0x0" ::: "memory");
  }

  // normalize, split and write head-major Y (hi/lo f16)
#pragma unroll
  for (int r = 0; r < 8; ++r) {
    const float inv = 1.0f / lrow[r];
    const size_t row = head_off + (size_t)(q0 + r + 8 * half) * HD;
#pragma unroll
    for (int s2 = 0; s2 < 4; ++s2) {
      const size_t addr = row + 16 * s2 + lq;
      split32(o[s2][r] * inv, &yh[addr], &yl[addr]);
    }
  }
}

// ---------------------------------------------------------------------------
// Phase 3: out = Y @ W_proj + b_proj.  A gathered from head-major Y hi/lo,
// B = W_proj^T hi/lo [768 x 768].  Wave: 64 x 16 strip.
// ---------------------------------------------------------------------------
__global__ __launch_bounds__(128) void proj_kernel(
    const half_t* __restrict__ yh, const half_t* __restrict__ yl,
    const half_t* __restrict__ wth, const half_t* __restrict__ wtl,
    const float* __restrict__ bp, float* __restrict__ out) {
  const int lane = threadIdx.x & 31;
  const int half = lane >> 4;
  const int lq = lane & 15;
  const int wave = blockIdx.x * (blockDim.x >> 5) + (threadIdx.x >> 5);
  const int nTilesN = D_MODEL / 16;  // 48
  const int nt = wave % nTilesN;
  const int ms = wave / nTilesN;
  const int m0 = ms * 64;
  const int n0 = nt * 16;
  const float bn = bp[n0 + lq];

  v8f acc[4] = {};
  for (int k0 = 0; k0 < D_MODEL; k0 += 32) {
    const int h = k0 >> 6;
    const int off = k0 & 63;
    const half_t* bp_h = wth + (size_t)(n0 + lq) * D_MODEL + k0 + 16 * half;
    const half_t* bp_l = wtl + (size_t)(n0 + lq) * D_MODEL + k0 + 16 * half;
    v16h bh = ldfrag(bp_h, bp_h + 8);
    v16h bl = ldfrag(bp_l, bp_l + 8);
#pragma unroll
    for (int s = 0; s < 4; ++s) {
      const int m = m0 + 16 * s + lq;
      const int bb = m >> 11;
      const int t = m & (SEQ_LEN - 1);
      const half_t* ap_h = yh +
          ((size_t)((bb * N_HEAD + h) * SEQ_LEN) + t) * HD + off + 16 * half;
      const half_t* ap_l = yl +
          ((size_t)((bb * N_HEAD + h) * SEQ_LEN) + t) * HD + off + 16 * half;
      v16h ah = ldfrag(ap_h, ap_h + 8);
      v16h al = ldfrag(ap_l, ap_l + 8);
      acc[s] = wmma16(ah, bh, acc[s]);
      acc[s] = wmma16(ah, bl, acc[s]);
      acc[s] = wmma16(al, bh, acc[s]);
    }
  }

#pragma unroll
  for (int s = 0; s < 4; ++s) {
#pragma unroll
    for (int r = 0; r < 8; ++r) {
      const int m = m0 + 16 * s + r + 8 * half;
      out[(size_t)m * D_MODEL + n0 + lq] = acc[s][r] + bn;
    }
  }
}

// ---------------------------------------------------------------------------
extern "C" void kernel_launch(void* const* d_in, const int* in_sizes, int n_in,
                              void* d_out, int out_size, void* d_ws,
                              size_t ws_size, hipStream_t stream) {
  (void)in_sizes;
  (void)n_in;
  (void)out_size;
  (void)ws_size;
  const float* x = (const float*)d_in[0];
  const float* W_attn = (const float*)d_in[1];
  const float* b_attn = (const float*)d_in[2];
  const float* W_proj = (const float*)d_in[3];
  const float* b_proj = (const float*)d_in[4];
  float* out = (float*)d_out;

  // Workspace layout (all _Float16).  x hi/lo region is reused for Y hi/lo
  // (x is dead after the QKV gemm; attention writes Y afterwards).
  half_t* p = (half_t*)d_ws;
  const size_t nX = (size_t)BATCH * SEQ_LEN * D_MODEL;   // 6,291,456
  const size_t nWA = (size_t)D_MODEL * N_QKV;            // 1,769,472
  const size_t nWP = (size_t)D_MODEL * D_MODEL;          //   589,824
  const size_t nQ = (size_t)BATCH * N_HEAD * SEQ_LEN * HD;  // == nX
  half_t* xh = p;            p += nX;
  half_t* xl = p;            p += nX;
  half_t* wah = p;           p += nWA;
  half_t* wal = p;           p += nWA;
  half_t* wph = p;           p += nWP;
  half_t* wpl = p;           p += nWP;
  half_t* qh = p;            p += nQ;
  half_t* ql = p;            p += nQ;
  half_t* kh = p;            p += nQ;
  half_t* kl = p;            p += nQ;
  half_t* vth = p;           p += nQ;
  half_t* vtl = p;           p += nQ;
  half_t* yh = xh;  // alias: x dead after phase 1
  half_t* yl = xl;

  // Phase 0: split / transpose-split prep
  split_kernel<<<dim3(2048), dim3(256), 0, stream>>>(x, xh, xl, (long)nX);
  transpose_split_kernel<<<dim3(2048), dim3(256), 0, stream>>>(
      W_attn, wah, wal, D_MODEL, N_QKV);
  transpose_split_kernel<<<dim3(1024), dim3(256), 0, stream>>>(
      W_proj, wph, wpl, D_MODEL, D_MODEL);

  // Phase 1: (8192/64) * (2304/16) = 18432 waves, 4 waves/block
  {
    const int waves = (BATCH * SEQ_LEN / 64) * (N_QKV / 16);
    qkv_kernel<<<dim3(waves / 4), dim3(128), 0, stream>>>(
        xh, xl, wah, wal, b_attn, qh, ql, kh, kl, vth, vtl);
  }
  // Phase 2: B*H*(T/16) = 6144 waves, 4 waves/block
  {
    const int waves = BATCH * N_HEAD * (SEQ_LEN / 16);
    attn_kernel<<<dim3(waves / FA_WAVES), dim3(32 * FA_WAVES), 0, stream>>>(
        qh, ql, kh, kl, vth, vtl, yh, yl);
  }
  // Phase 3: (8192/64) * (768/16) = 6144 waves, 4 waves/block
  {
    const int waves = (BATCH * SEQ_LEN / 64) * (D_MODEL / 16);
    proj_kernel<<<dim3(waves / 4), dim3(128), 0, stream>>>(yh, yl, wph, wpl,
                                                           b_proj, out);
  }
}
